// PretrainGRACE_24369644437904
// MI455X (gfx1250) — matrix operand order repaired and checked by reading, not compiled, not run on hardware
//
#include <hip/hip_runtime.h>

// ---------------------------------------------------------------------------
// GRACE loss on MI455X (gfx1250): all GEMMs via v_wmma_f32_16x16x32_bf16 with
// hi/lo bf16 error-compensated split (3 WMMAs ~= fp32 accuracy), similarity
// matrices fused with exp/rowsum epilogue (never materialized), A-band staged
// in LDS, deterministic fixed-order reductions (no float atomics).
// ---------------------------------------------------------------------------

#define HDIM   512
#define NROWS  8192
#define KSTEP  32
#define INV_TAU 2.0f                       // 1/0.5
#define EXP_SCALE 2.8853900817779268f      // (1/tau) * log2(e)

typedef __attribute__((ext_vector_type(4)))  unsigned int v4u;
typedef __attribute__((ext_vector_type(16))) __bf16       v16bf;
typedef __attribute__((ext_vector_type(8)))  float        v8f;

union V16 { v4u h[2]; v16bf b; };   // 32 bytes: two 16B chunks -> one WMMA operand

__device__ inline unsigned short f2bf(float f) {      // RNE float -> bf16 bits
  unsigned int u = __float_as_uint(f);
  u += 0x7FFFu + ((u >> 16) & 1u);
  return (unsigned short)(u >> 16);
}
__device__ inline float bf2f(unsigned short s) {
  return __uint_as_float(((unsigned int)s) << 16);
}
__device__ inline void split_store(float f, unsigned short* __restrict__ hi,
                                   unsigned short* __restrict__ lo, size_t idx) {
  unsigned short h = f2bf(f);
  hi[idx] = h;
  lo[idx] = f2bf(f - bf2f(h));
}

__device__ inline v8f wmma_bf16(v16bf a, v16bf b, v8f c) {
  return __builtin_amdgcn_wmma_f32_16x16x32_bf16(false, a, false, b, (short)0, c,
                                                 false, false);
}

// --------------------------- elementwise split ------------------------------
__global__ __launch_bounds__(256) void split_kernel(
    const float* __restrict__ in, unsigned short* __restrict__ hi,
    unsigned short* __restrict__ lo, int n, int do_relu) {
  int i = blockIdx.x * 256 + threadIdx.x;
  if (i >= n) return;
  float f = in[i];
  if (do_relu) f = fmaxf(f, 0.0f);
  split_store(f, hi, lo, (size_t)i);
}

// --------------------------- projector GEMM ---------------------------------
// C[i,j] = act( sum_k X[i,k]*W[j,k] + bias[j] )   (NT GEMM, torch Linear)
// block = 256 threads = 8 waves; block tile 64 rows x 64 cols.
// wave w: row group g=w&3 (16 rows), j half jt=w>>2 (two 16-col tiles).
// MODE 0: identity act, fp32 out.  MODE 1: ELU act, hi/lo split out.
template <int MODE>
__global__ __launch_bounds__(256) void proj_kernel(
    const unsigned short* __restrict__ Xhi, const unsigned short* __restrict__ Xlo,
    const unsigned short* __restrict__ Whi, const unsigned short* __restrict__ Wlo,
    const float* __restrict__ bias,
    unsigned short* __restrict__ Ohi, unsigned short* __restrict__ Olo,
    float* __restrict__ Of) {
  const int tid  = threadIdx.x;
  const int w    = tid >> 5, lane = tid & 31;
  const int half = lane >> 4, l16 = lane & 15;
  const int g    = w & 3, jt = w >> 2;
  const int m0   = blockIdx.y * 64 + 16 * g;
  const int j0   = blockIdx.x * 64 + 32 * jt;

  v8f acc0 = {0,0,0,0,0,0,0,0}, acc1 = {0,0,0,0,0,0,0,0};
  const size_t arow  = (size_t)(m0 + l16) * HDIM;
  const size_t brow0 = (size_t)(j0 + l16) * HDIM;
  const size_t brow1 = (size_t)(j0 + 16 + l16) * HDIM;

  for (int k0 = 0; k0 < HDIM; k0 += KSTEP) {
    V16 ah, al, bh, bl;
    // A operand: lanes<16: K {k0..k0+15},
    // lanes>=16: K {k0+16..k0+31}
    const v4u* pah = (const v4u*)(Xhi + arow + k0 + 16 * half);
    const v4u* pal = (const v4u*)(Xlo + arow + k0 + 16 * half);
    ah.h[0] = pah[0]; ah.h[1] = pah[1];
    al.h[0] = pal[0]; al.h[1] = pal[1];
    // B operand: lane holds 16 contiguous K of row (=output col), halves split K
    const v4u* pbh = (const v4u*)(Whi + brow0 + k0 + 16 * half);
    const v4u* pbl = (const v4u*)(Wlo + brow0 + k0 + 16 * half);
    bh.h[0] = pbh[0]; bh.h[1] = pbh[1];
    bl.h[0] = pbl[0]; bl.h[1] = pbl[1];
    acc0 = wmma_bf16(ah.b, bh.b, acc0);
    acc0 = wmma_bf16(ah.b, bl.b, acc0);
    acc0 = wmma_bf16(al.b, bh.b, acc0);
    pbh = (const v4u*)(Whi + brow1 + k0 + 16 * half);
    pbl = (const v4u*)(Wlo + brow1 + k0 + 16 * half);
    bh.h[0] = pbh[0]; bh.h[1] = pbh[1];
    bl.h[0] = pbl[0]; bl.h[1] = pbl[1];
    acc1 = wmma_bf16(ah.b, bh.b, acc1);
    acc1 = wmma_bf16(ah.b, bl.b, acc1);
    acc1 = wmma_bf16(al.b, bh.b, acc1);
  }

  for (int t = 0; t < 2; ++t) {
    v8f acc = t ? acc1 : acc0;
    int col = j0 + 16 * t + l16;
    float bv = bias[col];
    for (int r = 0; r < 8; ++r) {
      int row = m0 + r + 8 * half;           // C layout: VGPR r -> M = r + 8*half
      float v = acc[r] + bv;
      if (MODE == 1) {
        v = (v > 0.0f) ? v : (__builtin_amdgcn_exp2f(v * 1.44269504f) - 1.0f); // ELU
        split_store(v, Ohi, Olo, (size_t)row * HDIM + col);
      } else {
        Of[(size_t)row * HDIM + col] = v;
      }
    }
  }
}

// --------------------- row L2-normalize + hi/lo split -----------------------
__global__ __launch_bounds__(128) void norm_split_kernel(
    const float* __restrict__ h, unsigned short* __restrict__ hi,
    unsigned short* __restrict__ lo) {
  const int row = blockIdx.x, tid = threadIdx.x;
  const float* p = h + (size_t)row * HDIM;
  float s = 0.0f;
  for (int i = tid; i < HDIM; i += 128) { float v = p[i]; s += v * v; }
  for (int m = 16; m >= 1; m >>= 1) s += __shfl_xor(s, m, 32);
  __shared__ float ws4[4];
  if ((tid & 31) == 0) ws4[tid >> 5] = s;
  __syncthreads();
  float tot = ws4[0] + ws4[1] + ws4[2] + ws4[3];
  float inv = 1.0f / fmaxf(sqrtf(tot), 1e-12f);
  for (int i = tid; i < HDIM; i += 128)
    split_store(p[i] * inv, hi, lo, (size_t)row * HDIM + i);
}

// ---------------- fused similarity: rowsum(exp(X@Y^T / tau)) ----------------
// One block owns a 32-row band; loops over all 8192 cols in 64-col chunks.
// 8 waves: g=w&1 row group (16 rows), jt=w>>1 col tile (16 cols of chunk).
// A band (hi+lo) staged once in LDS (64 KB). Diagonal S values captured raw.
__global__ __launch_bounds__(256) void sim_kernel(
    const unsigned short* __restrict__ Xhi, const unsigned short* __restrict__ Xlo,
    const unsigned short* __restrict__ Yhi, const unsigned short* __restrict__ Ylo,
    float* __restrict__ rowsum, float* __restrict__ diag) {
  extern __shared__ char smem[];
  unsigned short* sAhi = (unsigned short*)smem;        // 32*512 bf16
  unsigned short* sAlo = sAhi + 32 * HDIM;             // 32*512 bf16
  float* sred = (float*)(sAlo + 32 * HDIM);            // 8 waves * 16 rows

  const int tid  = threadIdx.x;
  const int w    = tid >> 5, lane = tid & 31;
  const int half = lane >> 4, l16 = lane & 15;
  const int g    = w & 1, jt = w >> 1;
  const int m0   = blockIdx.x * 32;

  { // stage A band into LDS (global_load_b128 -> ds_store_b128)
    const v4u* gh = (const v4u*)(Xhi + (size_t)m0 * HDIM);
    const v4u* gl = (const v4u*)(Xlo + (size_t)m0 * HDIM);
    v4u* dh = (v4u*)sAhi; v4u* dl = (v4u*)sAlo;
    for (int i = tid; i < 32 * HDIM / 8; i += 256) { dh[i] = gh[i]; dl[i] = gl[i]; }
  }
  __syncthreads();

  float rs[8] = {0,0,0,0,0,0,0,0};
  const int arow = 16 * g + l16;
  const unsigned short* aH = sAhi + arow * HDIM;
  const unsigned short* aL = sAlo + arow * HDIM;

  for (int jc = 0; jc < NROWS; jc += 64) {
    const int jb = jc + 16 * jt;
    const size_t brow = (size_t)(jb + l16) * HDIM;
    v8f acc = {0,0,0,0,0,0,0,0};
    for (int k0 = 0; k0 < HDIM; k0 += KSTEP) {
      V16 ah, al, bh, bl;
      const v4u* pah = (const v4u*)(aH + k0 + 16 * half);
      const v4u* pal = (const v4u*)(aL + k0 + 16 * half);
      ah.h[0] = pah[0]; ah.h[1] = pah[1];
      al.h[0] = pal[0]; al.h[1] = pal[1];
      const v4u* pbh = (const v4u*)(Yhi + brow + k0 + 16 * half);
      const v4u* pbl = (const v4u*)(Ylo + brow + k0 + 16 * half);
      bh.h[0] = pbh[0]; bh.h[1] = pbh[1];
      bl.h[0] = pbl[0]; bl.h[1] = pbl[1];
      acc = wmma_bf16(ah.b, bh.b, acc);   // hi*hi
      acc = wmma_bf16(ah.b, bl.b, acc);   // hi*lo
      acc = wmma_bf16(al.b, bh.b, acc);   // lo*hi
    }
    const int col = jb + l16;
    for (int r = 0; r < 8; ++r) {
      float s = acc[r];
      float e = __builtin_amdgcn_exp2f(s * EXP_SCALE);  // exp(s/tau) via v_exp_f32
      rs[r] += e;
      int row = m0 + 16 * g + r + 8 * half;
      if (col == row) diag[row] = s;                    // raw similarity on diagonal
    }
  }

  // reduce partial rowsums across the 16 lanes of each half (fixed order)
  for (int r = 0; r < 8; ++r)
    for (int m = 8; m >= 1; m >>= 1) rs[r] += __shfl_xor(rs[r], m, 16);
  if (l16 == 0)
    for (int r = 0; r < 8; ++r) sred[w * 16 + 8 * half + r] = rs[r];
  __syncthreads();
  if (tid < 32) {                    // combine the 4 col-tile waves per row group
    int gg = tid >> 4, r = tid & 15;
    float v = 0.0f;
    for (int q = 0; q < 4; ++q) v += sred[(2 * q + gg) * 16 + r];
    rowsum[m0 + tid] = v;
  }
}

// --------------------------- final scalar loss ------------------------------
__global__ __launch_bounds__(256) void loss_kernel(
    const float* __restrict__ rs_aa, const float* __restrict__ rs_bb,
    const float* __restrict__ rs_ab, const float* __restrict__ rs_ba,
    const float* __restrict__ d_aa, const float* __restrict__ d_bb,
    const float* __restrict__ d_ab, const float* __restrict__ d_ba,
    float* __restrict__ out) {
  int tid = threadIdx.x;
  float acc = 0.0f;
  for (int i = tid; i < NROWS; i += 256) {
    float eaa = __builtin_amdgcn_exp2f(d_aa[i] * EXP_SCALE);
    float ebb = __builtin_amdgcn_exp2f(d_bb[i] * EXP_SCALE);
    float den1 = rs_aa[i] + rs_ab[i] - eaa;
    float den2 = rs_bb[i] + rs_ba[i] - ebb;
    float l1 = __logf(den1) - d_ab[i] * INV_TAU;   // -log(exp(d/tau)/den)
    float l2 = __logf(den2) - d_ba[i] * INV_TAU;
    acc += 0.5f * (l1 + l2);
  }
  __shared__ float red[256];
  red[tid] = acc;
  __syncthreads();
  for (int s = 128; s > 0; s >>= 1) {
    if (tid < s) red[tid] += red[tid + s];
    __syncthreads();
  }
  if (tid == 0) out[0] = red[0] / (float)NROWS;
}

// ----------------------------------------------------------------------------
extern "C" void kernel_launch(void* const* d_in, const int* in_sizes, int n_in,
                              void* d_out, int out_size, void* d_ws, size_t ws_size,
                              hipStream_t stream) {
  const float* z1 = (const float*)d_in[0];
  const float* z2 = (const float*)d_in[1];
  const float* W1 = (const float*)d_in[2];
  const float* b1 = (const float*)d_in[3];
  const float* W2 = (const float*)d_in[4];
  const float* b2 = (const float*)d_in[5];
  float* out = (float*)d_out;

  char* p = (char*)d_ws;
  auto alloc = [&](size_t bytes) -> char* {
    char* r = p;
    p += (bytes + 255) & ~(size_t)255;
    return r;
  };
  const size_t NH = (size_t)NROWS * HDIM;      // 4 Mi elements
  const size_t WH = (size_t)HDIM * HDIM;       // 256 Ki elements

  unsigned short* w1h = (unsigned short*)alloc(WH * 2);
  unsigned short* w1l = (unsigned short*)alloc(WH * 2);
  unsigned short* w2h = (unsigned short*)alloc(WH * 2);
  unsigned short* w2l = (unsigned short*)alloc(WH * 2);
  unsigned short* s0h = (unsigned short*)alloc(NH * 2);  // xs, later a_hi
  unsigned short* s0l = (unsigned short*)alloc(NH * 2);  // xs, later a_lo
  unsigned short* s1h = (unsigned short*)alloc(NH * 2);  // t,  later b_hi
  unsigned short* s1l = (unsigned short*)alloc(NH * 2);  // t,  later b_lo
  float* h1 = (float*)alloc(NH * 4);
  float* h2 = (float*)alloc(NH * 4);
  float* rs_aa = (float*)alloc(NROWS * 4);
  float* rs_bb = (float*)alloc(NROWS * 4);
  float* rs_ab = (float*)alloc(NROWS * 4);
  float* rs_ba = (float*)alloc(NROWS * 4);
  float* d_aa  = (float*)alloc(NROWS * 4);
  float* d_bb  = (float*)alloc(NROWS * 4);
  float* d_ab  = (float*)alloc(NROWS * 4);
  float* d_ba  = (float*)alloc(NROWS * 4);

  const dim3 projGrid(HDIM / 64, NROWS / 64);  // (8, 128)
  const size_t simSmem = 32 * HDIM * 2 * 2 + 8 * 16 * 4; // 66048 B

  // weight splits
  split_kernel<<<(int)((WH + 255) / 256), 256, 0, stream>>>(W1, w1h, w1l, (int)WH, 0);
  split_kernel<<<(int)((WH + 255) / 256), 256, 0, stream>>>(W2, w2h, w2l, (int)WH, 0);

  // view 1: relu+split -> layer1 (ELU, split out) -> layer2 (fp32 out)
  split_kernel<<<(int)((NH + 255) / 256), 256, 0, stream>>>(z1, s0h, s0l, (int)NH, 1);
  proj_kernel<1><<<projGrid, 256, 0, stream>>>(s0h, s0l, w1h, w1l, b1, s1h, s1l, nullptr);
  proj_kernel<0><<<projGrid, 256, 0, stream>>>(s1h, s1l, w2h, w2l, b2, nullptr, nullptr, h1);

  // view 2
  split_kernel<<<(int)((NH + 255) / 256), 256, 0, stream>>>(z2, s0h, s0l, (int)NH, 1);
  proj_kernel<1><<<projGrid, 256, 0, stream>>>(s0h, s0l, w1h, w1l, b1, s1h, s1l, nullptr);
  proj_kernel<0><<<projGrid, 256, 0, stream>>>(s1h, s1l, w2h, w2l, b2, nullptr, nullptr, h2);

  // normalize rows; a -> s0, b -> s1 (buffers reused)
  norm_split_kernel<<<NROWS, 128, 0, stream>>>(h1, s0h, s0l);
  norm_split_kernel<<<NROWS, 128, 0, stream>>>(h2, s1h, s1l);

  // fused similarity row-sums (+diagonals); (b,a) gives colsum of (a,b)
  sim_kernel<<<NROWS / 32, 256, simSmem, stream>>>(s0h, s0l, s0h, s0l, rs_aa, d_aa);
  sim_kernel<<<NROWS / 32, 256, simSmem, stream>>>(s1h, s1l, s1h, s1l, rs_bb, d_bb);
  sim_kernel<<<NROWS / 32, 256, simSmem, stream>>>(s0h, s0l, s1h, s1l, rs_ab, d_ab);
  sim_kernel<<<NROWS / 32, 256, simSmem, stream>>>(s1h, s1l, s0h, s0l, rs_ba, d_ba);

  loss_kernel<<<1, 256, 0, stream>>>(rs_aa, rs_bb, rs_ab, rs_ba, d_aa, d_bb, d_ab, d_ba, out);
}
